// SelfAttention_85633057948146
// MI455X (gfx1250) — matrix operand order repaired and checked by reading, not compiled, not run on hardware
//
#include <hip/hip_runtime.h>

#define BATCH 8
#define CH    256
#define NPOS  4096
#define CQ    32

typedef _Float16 v16h __attribute__((ext_vector_type(16)));
typedef _Float16 v8h  __attribute__((ext_vector_type(8)));
typedef _Float16 v4h  __attribute__((ext_vector_type(4)));
typedef float    v8f  __attribute__((ext_vector_type(8)));
typedef float    v4f  __attribute__((ext_vector_type(4)));

static __device__ __forceinline__ v16h cat8(v8h lo, v8h hi) {
  return __builtin_shufflevector(lo, hi, 0,1,2,3,4,5,6,7,8,9,10,11,12,13,14,15);
}

static __device__ __forceinline__ v8f wmma16(v16h a, v16h b, v8f c) {
  // v_wmma_f32_16x16x32_f16
  return __builtin_amdgcn_wmma_f32_16x16x32_f16(false, a, false, b, (short)0, c, false, false);
}

// ---------------------------------------------------------------------------
// Kernel 1: fused 1x1-conv projections q/k/v with WMMA.
// Grid: BATCH * (NPOS/64) blocks, 256 threads (8 waves).
// Outputs: Qh [B][N][32] f16, Kh [B][N][32] f16, Vt [B][C][N] f16.
// ---------------------------------------------------------------------------
__global__ __launch_bounds__(256) void proj_qkv_kernel(
    const float* __restrict__ x,
    const float* __restrict__ Wq, const float* __restrict__ bq,
    const float* __restrict__ Wk, const float* __restrict__ bk,
    const float* __restrict__ Wv, const float* __restrict__ bv,
    _Float16* __restrict__ Qh, _Float16* __restrict__ Kh,
    _Float16* __restrict__ Vt)
{
  __shared__ _Float16 xT[64 * 264];   // [n_local][i], row stride 264 halfs (528B, 16B aligned)

  const int t  = threadIdx.x;
  const int b  = blockIdx.x >> 6;          // 64 n-blocks per batch
  const int n0 = (blockIdx.x & 63) * 64;
  const float* xb = x + (size_t)b * CH * NPOS;

  // Stage x[b][:, n0:n0+64] into LDS transposed as f16.
  for (int s = 0; s < 16; ++s) {
    int idx = t + 256 * s;                 // 4096 float4 tiles
    int i   = idx >> 4;                    // channel row 0..255
    int n4  = (idx & 15) * 4;              // col in block
    v4f v = *(const v4f*)(xb + (size_t)i * NPOS + n0 + n4);
#pragma unroll
    for (int j = 0; j < 4; ++j)
      xT[(n4 + j) * 264 + i] = (_Float16)v[j];
  }
  __syncthreads();

  const int lane  = t & 31;
  const int col   = lane & 15;
  const int hi    = lane >> 4;
  const int akoff = hi ? 8 : 0;            // A-operand half-lane K interleave
  const int bkoff = hi ? 8 : 0;            // B-operand half-lane K interleave
  // Wave id as a proven-scalar value so all rt-dependent branches are SALU.
  const int wu = __builtin_amdgcn_readfirstlane(t >> 5);

  // 20 row-tiles of 16 output rows: 0-1 q, 2-3 k, 4-19 v. Wave w takes rt = w, w+8, w+16.
  for (int rt = wu; rt < 20; rt += 8) {
    const float* Wm; const float* bias; int row0;
    if (rt < 2)      { Wm = Wq; bias = bq; row0 = rt * 16; }
    else if (rt < 4) { Wm = Wk; bias = bk; row0 = (rt - 2) * 16; }
    else             { Wm = Wv; bias = bv; row0 = (rt - 4) * 16; }
    const int orow = row0 + col;

    // Load the 8 K-block A tiles (f32 weights -> f16) once, reuse for 4 n-subtiles.
    v16h aW[8];
#pragma unroll
    for (int kb = 0; kb < 8; ++kb) {
      const float* wp = Wm + (size_t)orow * CH + kb * 32 + akoff;
      v4f w0 = *(const v4f*)(wp);
      v4f w1 = *(const v4f*)(wp + 4);
      v4f w2 = *(const v4f*)(wp + 16);
      v4f w3 = *(const v4f*)(wp + 20);
      v16h a;
#pragma unroll
      for (int j = 0; j < 4; ++j) {
        a[j]      = (_Float16)w0[j];
        a[4 + j]  = (_Float16)w1[j];
        a[8 + j]  = (_Float16)w2[j];
        a[12 + j] = (_Float16)w3[j];
      }
      aW[kb] = a;
    }

    float bias8[8];
#pragma unroll
    for (int i = 0; i < 8; ++i) bias8[i] = bias[row0 + i + 8 * hi];

    for (int ns = 0; ns < 4; ++ns) {
      v8f acc = {};
      const _Float16* xrow = &xT[(ns * 16 + col) * 264];
#pragma unroll
      for (int kb = 0; kb < 8; ++kb) {
        v8h b0 = *(const v8h*)(xrow + kb * 32 + bkoff);
        v8h b1 = *(const v8h*)(xrow + kb * 32 + bkoff + 16);
        acc = wmma16(aW[kb], cat8(b0, b1), acc);
      }
      const int n = n0 + ns * 16 + col;    // D-layout: lane = output column = n
      if (rt < 4) {                        // scalar branch (rt derived from readfirstlane)
        _Float16* dst = (rt < 2) ? Qh : Kh;
#pragma unroll
        for (int i = 0; i < 8; ++i) {
          const int r = row0 + i + 8 * hi; // D-layout: VGPR i = output row
          dst[((size_t)b * NPOS + n) * CQ + r] = (_Float16)(acc[i] + bias8[i]);
        }
      } else {
#pragma unroll
        for (int i = 0; i < 8; ++i) {
          const int r = row0 + i + 8 * hi;
          Vt[((size_t)b * CH + r) * NPOS + n] = (_Float16)(acc[i] + bias8[i]);
        }
      }
    }
  }
}

// ---------------------------------------------------------------------------
// Kernel 2: flash attention + gamma*out + x, with double-buffered K/V staging.
// Grid: BATCH * 32 blocks, 256 threads (8 waves). Each wave owns 16 query rows.
// ---------------------------------------------------------------------------
__global__ __launch_bounds__(256) void flash_attn_kernel(
    const _Float16* __restrict__ Qh, const _Float16* __restrict__ Kh,
    const _Float16* __restrict__ Vt, const float* __restrict__ x,
    const float* __restrict__ gamma, float* __restrict__ out)
{
  __shared__ _Float16 Ks[2][32 * 40];     // 32 keys x 32 d, stride 40 halfs (80B)
  __shared__ _Float16 Vs[2][256 * 40];    // 256 ch x 32 keys
  __shared__ _Float16 Ps[8][16 * 40];     // per-wave P tile (16 m x 32 keys)
  __shared__ float    Tf[8][16];          // per-wave row->lane broadcast scratch

  const int t     = threadIdx.x;
  const int wave  = t >> 5;
  const int lane  = t & 31;
  const int col   = lane & 15;
  const int hi    = lane >> 4;
  const int akoff = hi ? 8 : 0;
  const int bkoff = hi ? 8 : 0;

  const int b  = blockIdx.x >> 5;
  const int m0 = ((blockIdx.x & 31) * 8 + wave) * 16;

  // Q tile as A operand (rows = queries m0..m0+15).
  const _Float16* qrow = Qh + ((size_t)b * NPOS + m0 + col) * CQ;
  const v16h aQ = cat8(*(const v8h*)(qrow + akoff), *(const v8h*)(qrow + akoff + 16));

  float mrow[8], lrow[8];
  v8f O[16];
#pragma unroll
  for (int i = 0; i < 8; ++i) { mrow[i] = -__builtin_inff(); lrow[i] = 0.0f; }
#pragma unroll
  for (int cg = 0; cg < 16; ++cg) O[cg] = (v8f){};

  const _Float16* Kb = Kh + (size_t)b * NPOS * CQ;
  const _Float16* Vb = Vt + (size_t)b * CH * NPOS;
  const float LOG2E = 1.44269504088896340736f;
  _Float16* pw = &Ps[wave][0];
  float*    Tw = &Tf[wave][0];

  // Per-thread staging assignment.
  const int sr  = t >> 3;                 // K stage: row (key) 0..31
  const int sd4 = (t & 7) * 4;            // K stage: 4-half chunk
  const _Float16* vrow = Vb + (size_t)t * NPOS;  // V stage: channel row = t

  // Prologue: stage key-block 0 into buffer 0.
  {
    v4h kreg = *(const v4h*)(Kb + (size_t)sr * CQ + sd4);
    v8h vreg[4];
#pragma unroll
    for (int j = 0; j < 4; ++j) vreg[j] = *(const v8h*)(vrow + j * 8);
    *(v4h*)(&Ks[0][sr * 40 + sd4]) = kreg;
#pragma unroll
    for (int j = 0; j < 4; ++j) *(v8h*)(&Vs[0][t * 40 + j * 8]) = vreg[j];
  }

  for (int kb = 0; kb < 128; ++kb) {
    __syncthreads();                      // buffer (kb&1) published to all waves
    const int buf = kb & 1;

    // Issue next block's global loads now; they complete under LOADcnt while
    // this block's WMMAs run.
    v4h kreg;
    v8h vreg[4];
    if (kb + 1 < 128) {
      kreg = *(const v4h*)(Kb + ((size_t)((kb + 1) * 32 + sr)) * CQ + sd4);
#pragma unroll
      for (int j = 0; j < 4; ++j)
        vreg[j] = *(const v8h*)(vrow + (kb + 1) * 32 + j * 8);
    }

    // ---- S = Q * K^T (two 16-key tiles): rows(m) in VGPRs, cols(key) in lanes ----
    const _Float16* kr0 = &Ks[buf][col * 40 + bkoff];
    const _Float16* kr1 = &Ks[buf][(col + 16) * 40 + bkoff];
    v8f S0 = wmma16(aQ, cat8(*(const v8h*)kr0, *(const v8h*)(kr0 + 16)), (v8f){});
    v8f S1 = wmma16(aQ, cat8(*(const v8h*)kr1, *(const v8h*)(kr1 + 16)), (v8f){});

    // ---- online softmax in row(=VGPR)/col(=lane) space ----
    float sc[8];
#pragma unroll
    for (int i = 0; i < 8; ++i) {
      float tmax = fmaxf(S0[i], S1[i]);
#pragma unroll
      for (int d = 1; d < 16; d <<= 1)
        tmax = fmaxf(tmax, __shfl_xor(tmax, d, 32));
      float mn = fmaxf(mrow[i], tmax);
      sc[i] = __builtin_amdgcn_exp2f((mrow[i] - mn) * LOG2E);
      float p0 = __builtin_amdgcn_exp2f((S0[i] - mn) * LOG2E);
      float p1 = __builtin_amdgcn_exp2f((S1[i] - mn) * LOG2E);
      S0[i] = p0; S1[i] = p1;
      float rs = p0 + p1;
#pragma unroll
      for (int d = 1; d < 16; d <<= 1)
        rs += __shfl_xor(rs, d, 32);
      lrow[i] = lrow[i] * sc[i] + rs;
      mrow[i] = mn;
    }

    // ---- bounce P (C-layout f32 -> f16 [m][key]) and the per-m rescale factors
    //      through per-wave LDS; one dscnt wait covers both ----
#pragma unroll
    for (int i = 0; i < 8; ++i) {
      pw[(i + 8 * hi) * 40 + col]      = (_Float16)S0[i];
      pw[(i + 8 * hi) * 40 + col + 16] = (_Float16)S1[i];
    }
    if (col == 0) {                     // lane 0 -> rows 0..7, lane 16 -> rows 8..15
#pragma unroll
      for (int i = 0; i < 8; ++i) Tw[i + 8 * hi] = sc[i];
    }
    asm volatile("s_wait_dscnt 0x0" ::: "memory");

    // B operand P^T: lane col = query m, K dim = key index (contiguous in row m).
    const _Float16* pr = &pw[col * 40 + bkoff];
    const v16h bP = cat8(*(const v8h*)pr, *(const v8h*)(pr + 16));
    // Per-lane rescale factor for O (O cols = m = lane).
    const float sc_lane = Tw[col];

    // ---- rescale accumulators, then O += V * P^T (rows = channels, cols = m) ----
#pragma unroll
    for (int cg = 0; cg < 16; ++cg) {
#pragma unroll
      for (int i = 0; i < 8; ++i) O[cg][i] *= sc_lane;
    }
#pragma unroll
    for (int cg = 0; cg < 16; ++cg) {
      const _Float16* vr = &Vs[buf][(cg * 16 + col) * 40 + akoff];
      v16h aV = cat8(*(const v8h*)vr, *(const v8h*)(vr + 16));
      O[cg] = wmma16(aV, bP, O[cg]);
    }

    // Write next block into the other buffer (its consumers sit behind the
    // next __syncthreads, which also waits out these DS stores).
    if (kb + 1 < 128) {
      *(v4h*)(&Ks[buf ^ 1][sr * 40 + sd4]) = kreg;
#pragma unroll
      for (int j = 0; j < 4; ++j)
        *(v8h*)(&Vs[buf ^ 1][t * 40 + j * 8]) = vreg[j];
    }
  }

  // ---- finalize: out = gamma * (O / l) + x. O rows = channels (VGPR),
  //      cols = m (lane) -> stores are coalesced over m. 1/l must follow
  //      the lane dim, so broadcast it through LDS like sc. ----
  const float g = gamma[0];
  if (col == 0) {
#pragma unroll
    for (int i = 0; i < 8; ++i) Tw[i + 8 * hi] = 1.0f / lrow[i];
  }
  asm volatile("s_wait_dscnt 0x0" ::: "memory");
  const float linv_lane = Tw[col];

  const float* xb = x + (size_t)b * CH * NPOS;
  float* ob = out + (size_t)b * CH * NPOS;
#pragma unroll
  for (int cg = 0; cg < 16; ++cg) {
#pragma unroll
    for (int i = 0; i < 8; ++i) {
      const int c = cg * 16 + i + 8 * hi;              // O row (VGPR) = channel
      const size_t idx = (size_t)c * NPOS + m0 + col;  // lane = m -> coalesced
      ob[idx] = g * (O[cg][i] * linv_lane) + xb[idx];
    }
  }
}

// ---------------------------------------------------------------------------
extern "C" void kernel_launch(void* const* d_in, const int* in_sizes, int n_in,
                              void* d_out, int out_size, void* d_ws, size_t ws_size,
                              hipStream_t stream) {
  (void)in_sizes; (void)n_in; (void)out_size; (void)ws_size;
  const float* x     = (const float*)d_in[0];
  const float* Wq    = (const float*)d_in[1];
  const float* bq    = (const float*)d_in[2];
  const float* Wk    = (const float*)d_in[3];
  const float* bk    = (const float*)d_in[4];
  const float* Wv    = (const float*)d_in[5];
  const float* bv    = (const float*)d_in[6];
  const float* gamma = (const float*)d_in[7];
  float* out = (float*)d_out;

  _Float16* Qh = (_Float16*)d_ws;                       // B*N*32 f16
  _Float16* Kh = Qh + (size_t)BATCH * NPOS * CQ;        // B*N*32 f16
  _Float16* Vt = Kh + (size_t)BATCH * NPOS * CQ;        // B*C*N  f16

  proj_qkv_kernel<<<BATCH * (NPOS / 64), 256, 0, stream>>>(
      x, Wq, bq, Wk, bk, Wv, bv, Qh, Kh, Vt);
  flash_attn_kernel<<<BATCH * 32, 256, 0, stream>>>(
      Qh, Kh, Vt, x, gamma, out);
}
